// MambaListOpsClassifierLuenberger_46145128628829
// MI455X (gfx1250) — matrix operand order repaired and checked by reading, not compiled, not run on hardware
//
#include <hip/hip_runtime.h>

// ---------------- problem constants (from the reference) ----------------
#define Bsz    2
#define Lseq   1024
#define BL     (Bsz * Lseq)      // 2048 tokens
#define DM     256               // d_model
#define NLAY   6
#define DI     512               // inner dim
#define DS     16                // state dim
#define DC     4                 // conv taps
#define DTR    16                // dt rank
#define NC     10                // classes
#define ALPHA  0.1f

// ---------------- CDNA5 WMMA / descriptor types ----------------
typedef __attribute__((ext_vector_type(16))) __bf16 v16bf;
typedef __attribute__((ext_vector_type(8)))  float        v8f;
typedef __attribute__((ext_vector_type(8)))  unsigned int v8u;
typedef __attribute__((ext_vector_type(4)))  unsigned int v4u;

union Frag { v8u u; v16bf h; };

// round-to-nearest-even f32 -> bf16 (as raw u16)
__device__ __forceinline__ unsigned short f2bf(float f) {
    unsigned int u = __float_as_uint(f);
    u += 0x7FFFu + ((u >> 16) & 1u);
    return (unsigned short)(u >> 16);
}

__device__ __forceinline__ float silu(float x) {
    return x / (1.0f + __expf(-x));
}

// ---------------- f32 -> bf16 weight conversion ----------------
__global__ void cvt_f32_bf16_kernel(const float* __restrict__ src,
                                    unsigned short* __restrict__ dst, int n) {
    int i = blockIdx.x * blockDim.x + threadIdx.x;
    if (i < n) dst[i] = f2bf(src[i]);
}

// ---------------- embedding + positional ----------------
__global__ void embed_pos_kernel(const int* __restrict__ tok,
                                 const float* __restrict__ emb,
                                 const float* __restrict__ pos,
                                 float* __restrict__ h) {
    int t = blockIdx.x * blockDim.x + threadIdx.x;
    if (t >= BL * DM) return;
    int bl = t / DM, d = t % DM;
    int l  = bl % Lseq;
    h[t] = emb[tok[bl] * DM + d] + pos[l * DM + d];
}

// ---------------- LayerNorm over 256-dim rows, one wave32 per row ------
__global__ void ln_rows_bf16_kernel(const float* __restrict__ h,
                                    const float* __restrict__ w,
                                    const float* __restrict__ b,
                                    unsigned short* __restrict__ out,
                                    int rows) {
    int gid  = blockIdx.x * blockDim.x + threadIdx.x;
    int wave = gid >> 5;
    int lane = threadIdx.x & 31;
    if (wave >= rows) return;
    const float* row = h + (size_t)wave * DM;
    float v[8], s = 0.0f;
#pragma unroll
    for (int j = 0; j < 8; ++j) { v[j] = row[lane + 32 * j]; s += v[j]; }
#pragma unroll
    for (int off = 16; off > 0; off >>= 1) s += __shfl_xor(s, off);
    float mean = s * (1.0f / DM);
    float var = 0.0f;
#pragma unroll
    for (int j = 0; j < 8; ++j) { float d0 = v[j] - mean; var += d0 * d0; }
#pragma unroll
    for (int off = 16; off > 0; off >>= 1) var += __shfl_xor(var, off);
    float inv = rsqrtf(var * (1.0f / DM) + 1e-5f);
#pragma unroll
    for (int j = 0; j < 8; ++j) {
        int c = lane + 32 * j;
        out[(size_t)wave * DM + c] = f2bf((v[j] - mean) * inv * w[c] + b[c]);
    }
}

// ---------------- bf16 WMMA GEMM: C[M,N] = A[M,K] * W[N,K]^T (+res) ----
// Each wave computes a 16 x (NT*16) output strip: one A fragment is
// amortized over NT v_wmma issues per K-step (WMMA-rate bound, operands
// are L2-resident on a 192MB-L2 part). A and B fragments both use contiguous
// k order (k = 16*half + element), not the ISA table order. The tile guard is
// wave-uniform so EXEC is all-ones at every v_wmma.
template<int NT>
__global__ void wmma_gemm_bf16_kernel(const unsigned short* __restrict__ A,
                                      const unsigned short* __restrict__ W,
                                      const float* __restrict__ residual,
                                      float* __restrict__ C,
                                      int M, int N, int K) {
    int wave     = blockIdx.x * (blockDim.x >> 5) + (threadIdx.x >> 5);
    int tiles_n  = N >> 4;
    int groups_n = tiles_n / NT;
    int tiles_m  = M >> 4;
    if (wave >= tiles_m * groups_n) return;           // whole-wave exit
    int tm = wave / groups_n, tg = wave % groups_n;

    int lane = threadIdx.x & 31;
    int hi   = lane >> 4;          // 0: lanes 0-15, 1: lanes 16-31
    int l15  = lane & 15;

    const unsigned short* arow  = A + (size_t)(tm * 16 + l15) * K;
    const unsigned short* wrow0 = W + (size_t)(tg * NT * 16 + l15) * K;

    v8f acc[NT];
#pragma unroll
    for (int nt = 0; nt < NT; ++nt) acc[nt] = {};

    for (int k0 = 0; k0 < K; k0 += 32) {
        __builtin_prefetch(arow + k0 + 32, 0, 1);     // -> global_prefetch_b8
        Frag a;
        // A 16x32 bf16: lanes<16 VGPR0-7 K=k0+0..15,
        //               lanes>=16 K=k0+16..31 (matches B).
        int kaLo = k0 + (hi ? 16 : 0);
        int kaHi = k0 + 8 + (hi ? 16 : 0);
#pragma unroll
        for (int j = 0; j < 4; ++j) {
            a.u[j]     = *(const unsigned int*)(arow + kaLo + 2 * j);
            a.u[4 + j] = *(const unsigned int*)(arow + kaHi + 2 * j);
        }
        // B 32x16 bf16: lanes<16 hold K=k0..k0+15 for N=lane,
        //               lanes>=16 hold K=k0+16..k0+31 for N=lane-16.
        int kb = k0 + (hi ? 16 : 0);
#pragma unroll
        for (int nt = 0; nt < NT; ++nt) {
            const unsigned short* wrow = wrow0 + (size_t)nt * 16 * K;
            Frag bfr;
#pragma unroll
            for (int j = 0; j < 8; ++j)
                bfr.u[j] = *(const unsigned int*)(wrow + kb + 2 * j);
            acc[nt] = __builtin_amdgcn_wmma_f32_16x16x32_bf16(
                /*neg_a=*/false, a.h, /*neg_b=*/false, bfr.h,
                /*c_mod=*/(short)0, acc[nt], /*reuse_a=*/false, /*reuse_b=*/false);
        }
    }

    // C/D f32 16x16: VGPR r is row m0+r (lanes<16) or m0+8+r (lanes>=16)
    int row0 = tm * 16 + hi * 8;
#pragma unroll
    for (int nt = 0; nt < NT; ++nt) {
        int col = (tg * NT + nt) * 16 + l15;
#pragma unroll
        for (int r = 0; r < 8; ++r) {
            size_t idx = (size_t)(row0 + r) * N + col;
            float v = acc[nt][r];
            if (residual) v += residual[idx];
            C[idx] = v;
        }
    }
}

// ---------------- depthwise causal conv(4) + SiLU ----------------------
__global__ void conv_silu_kernel(const float* __restrict__ xz,
                                 const float* __restrict__ cw,   // (DI,DC)
                                 const float* __restrict__ cb,   // (DI)
                                 float* __restrict__ xc,
                                 unsigned short* __restrict__ xc_bf) {
    int t = blockIdx.x * blockDim.x + threadIdx.x;
    if (t >= BL * DI) return;
    int bl = t / DI, d = t % DI;
    int b  = bl / Lseq, l = bl % Lseq;
    float acc = cb[d];
#pragma unroll
    for (int k = 0; k < DC; ++k) {
        int ls = l + k - (DC - 1);
        if (ls >= 0)
            acc += cw[d * DC + k] * xz[((size_t)(b * Lseq + ls)) * (2 * DI) + d];
    }
    float s = silu(acc);
    xc[t]    = s;
    xc_bf[t] = f2bf(s);
}

// ---------------- delta = softplus(dt @ dt_proj_w^T + b) ---------------
__global__ void dt_softplus_kernel(const float* __restrict__ xdbl,  // (BL,48)
                                   const float* __restrict__ dtw,   // (DI,DTR)
                                   const float* __restrict__ dtb,   // (DI)
                                   float* __restrict__ delta) {
    int t = blockIdx.x * blockDim.x + threadIdx.x;
    if (t >= BL * DI) return;
    int bl = t / DI, d = t % DI;
    const float* dtv  = xdbl + (size_t)bl * 48;
    const float* wrow = dtw + d * DTR;
    float acc = dtb[d];
#pragma unroll
    for (int j = 0; j < DTR; ++j) acc += dtv[j] * wrow[j];
    delta[t] = (acc > 20.0f) ? acc : log1pf(__expf(acc));
}

// ---------------- Luenberger selective scan ----------------------------
// The whole (L x 32) B/C time-series for this batch (128 KB) is staged
// into LDS with ONE Tensor Data Mover descriptor (2D tile: 32 wide,
// L rows, row stride 48 f32) -> tensor_load_to_lds + s_wait_tensorcnt.
// After a single barrier the L=1024 recurrence runs from registers + LDS
// with no per-timestep synchronization or global B/C traffic.
__global__ void scan_kernel(const float* __restrict__ delta, // (B,L,DI)
                            const float* __restrict__ xdbl,  // (B,L,48)
                            const float* __restrict__ xc,    // (B,L,DI)
                            const float* __restrict__ A_log, // (DI,DS)
                            float* __restrict__ y) {         // (B,L,DI)
    extern __shared__ float sBC[];                 // (Lseq, 32): [B(16) C(16)]
    int b = blockIdx.x / (DI / 256);
    int d = (blockIdx.x % (DI / 256)) * 256 + threadIdx.x;

    if (threadIdx.x == 0) {
        // ---- Tensor DMA descriptor (D#): groups 0 and 1 (2D tensor) ----
        unsigned long long ga =
            (unsigned long long)(const void*)(xdbl + (size_t)b * Lseq * 48 + DTR);
        unsigned ldsOff = (unsigned)(size_t)(void*)sBC;   // LDS byte offset
        v4u g0;
        g0[0] = 1u;                                    // count=1 valid descriptor
        g0[1] = ldsOff;                                // lds_addr [63:32]
        g0[2] = (unsigned)ga;                          // global_addr low  [95:64]
        g0[3] = ((unsigned)(ga >> 32) & 0x01FFFFFFu)   // global_addr high [120:96]
                | (2u << 30);                          // type=2 ("image") [127:126]
        v8u g1;
        g1[0] = (2u << 16);                 // workgroup_mask=0, data_size=2 (4B)
        g1[1] = 0xFFFF0000u;                // tensor_dim0 lo16 (large, no OOB)
        g1[2] = 0x0000FFFFu | 0xFFFF0000u;  // tensor_dim0 hi16 | tensor_dim1 lo16
        g1[3] = 0x0000FFFFu | (32u << 16);  // tensor_dim1 hi16 | tile_dim0=32
        g1[4] = (unsigned)Lseq;             // tile_dim1=L, tile_dim2=0
        g1[5] = 48u;                        // tensor_dim0_stride = 48 elements
        g1[6] = 0u;                         // stride hi / dim1_stride lo
        g1[7] = 0u;
        asm volatile("tensor_load_to_lds %0, %1" :: "s"(g0), "s"(g1) : "memory");
        __builtin_amdgcn_s_wait_tensorcnt(0);  // TDM ignores EXEC; done here
    }
    __syncthreads();

    float A[DS];
#pragma unroll
    for (int n = 0; n < DS; ++n) A[n] = -__expf(A_log[d * DS + n]);

    float hs[DS], hh[DS];
#pragma unroll
    for (int n = 0; n < DS; ++n) { hs[n] = 0.0f; hh[n] = 0.0f; }

    const float* drow = delta + (size_t)b * Lseq * DI + d;
    const float* urow = xc    + (size_t)b * Lseq * DI + d;
    float*       yrow = y     + (size_t)b * Lseq * DI + d;

    for (int t = 0; t < Lseq; ++t) {
        const float* bc = sBC + t * 32;              // [B(16) | C(16)]
        float dl = drow[(size_t)t * DI];
        float u  = urow[(size_t)t * DI];
        float acc = 0.0f;
#pragma unroll
        for (int n = 0; n < DS; ++n) {
            float da   = __expf(dl * A[n]);
            float dbu  = dl * bc[n] * u;
            hs[n]      = da * hs[n] + dbu;
            float pred = da * hh[n] + dbu;
            hh[n]      = pred + ALPHA * (hs[n] - pred);
            acc       += hh[n] * bc[DS + n];
        }
        yrow[(size_t)t * DI] = acc;
    }
}

// ---------------- y = (scan + Dp*xc) * silu(z), to bf16 ----------------
__global__ void gate_kernel(const float* __restrict__ y,
                            const float* __restrict__ xc,
                            const float* __restrict__ Dp,     // (DI)
                            const float* __restrict__ xz,     // z = cols DI..2DI
                            unsigned short* __restrict__ yg_bf) {
    int t = blockIdx.x * blockDim.x + threadIdx.x;
    if (t >= BL * DI) return;
    int bl = t / DI, d = t % DI;
    float z = xz[(size_t)bl * (2 * DI) + DI + d];
    float v = (y[t] + Dp[d] * xc[t]) * silu(z);
    yg_bf[t] = f2bf(v);
}

// ---------------- masked mean pool -------------------------------------
__global__ void pool_kernel(const float* __restrict__ h,
                            const int* __restrict__ mask,
                            float* __restrict__ pooled) {
    int t = blockIdx.x * blockDim.x + threadIdx.x;
    if (t >= Bsz * DM) return;
    int b = t / DM, d = t % DM;
    float s = 0.0f, cnt = 0.0f;
    for (int l = 0; l < Lseq; ++l) {
        float m = (float)mask[b * Lseq + l];
        s   += m * h[((size_t)(b * Lseq + l)) * DM + d];
        cnt += m;
    }
    pooled[t] = s / cnt;
}

// ---------------- final LN + classifier head ---------------------------
__global__ void final_head_kernel(const float* __restrict__ pooled,
                                  const float* __restrict__ nw,
                                  const float* __restrict__ nb,
                                  const float* __restrict__ cw,  // (NC,DM)
                                  const float* __restrict__ cb,  // (NC)
                                  float* __restrict__ out) {
    __shared__ float red[DM];
    __shared__ float lnv[DM];
    __shared__ float stats[2];
    int b = blockIdx.x, d = threadIdx.x;
    float v = pooled[b * DM + d];
    red[d] = v; __syncthreads();
    for (int s = DM / 2; s > 0; s >>= 1) { if (d < s) red[d] += red[d + s]; __syncthreads(); }
    if (d == 0) stats[0] = red[0] * (1.0f / DM);
    __syncthreads();
    float mean = stats[0];
    float dv = v - mean;
    red[d] = dv * dv; __syncthreads();
    for (int s = DM / 2; s > 0; s >>= 1) { if (d < s) red[d] += red[d + s]; __syncthreads(); }
    if (d == 0) stats[1] = rsqrtf(red[0] * (1.0f / DM) + 1e-5f);
    __syncthreads();
    lnv[d] = dv * stats[1] * nw[d] + nb[d];
    __syncthreads();
    if (d < NC) {
        float acc = cb[d];
        for (int j = 0; j < DM; ++j) acc += lnv[j] * cw[d * DM + j];
        out[b * NC + d] = acc;
    }
}

// =======================================================================
extern "C" void kernel_launch(void* const* d_in, const int* in_sizes, int n_in,
                              void* d_out, int out_size, void* d_ws, size_t ws_size,
                              hipStream_t stream) {
    const float* emb        = (const float*)d_in[0];
    const float* pos_emb    = (const float*)d_in[1];
    const float* ln_w       = (const float*)d_in[2];
    const float* ln_b       = (const float*)d_in[3];
    const float* in_proj_w  = (const float*)d_in[4];   // (NL, 2*DI, DM)
    const float* conv_w     = (const float*)d_in[5];   // (NL, DI, DC)
    const float* conv_b     = (const float*)d_in[6];   // (NL, DI)
    const float* x_proj_w   = (const float*)d_in[7];   // (NL, 48, DI)
    const float* dt_proj_w  = (const float*)d_in[8];   // (NL, DI, DTR)
    const float* dt_proj_b  = (const float*)d_in[9];   // (NL, DI)
    const float* A_log      = (const float*)d_in[10];  // (NL, DI, DS)
    const float* Dp         = (const float*)d_in[11];  // (NL, DI)
    const float* out_proj_w = (const float*)d_in[12];  // (NL, DM, DI)
    const float* norm_w     = (const float*)d_in[13];
    const float* norm_b     = (const float*)d_in[14];
    const float* cls_w      = (const float*)d_in[15];
    const float* cls_b      = (const float*)d_in[16];
    const int*   tok        = (const int*)d_in[17];
    const int*   mask       = (const int*)d_in[18];
    float*       out        = (float*)d_out;

    // -------- workspace bump allocator (256B aligned) --------
    char* wsb = (char*)d_ws;
    size_t off = 0;
    auto alloc = [&](size_t bytes) -> void* {
        void* p = wsb + off;
        off = (off + bytes + 255) & ~(size_t)255;
        return p;
    };
    unsigned short* w_in_bf = (unsigned short*)alloc((size_t)NLAY * 2 * DI * DM * 2);
    unsigned short* w_xp_bf = (unsigned short*)alloc((size_t)NLAY * 48 * DI * 2);
    unsigned short* w_op_bf = (unsigned short*)alloc((size_t)NLAY * DM * DI * 2);
    float* hA               = (float*)alloc((size_t)BL * DM * 4);
    float* hB               = (float*)alloc((size_t)BL * DM * 4);
    unsigned short* hn_bf   = (unsigned short*)alloc((size_t)BL * DM * 2);
    float* xz               = (float*)alloc((size_t)BL * 2 * DI * 4);
    float* xc               = (float*)alloc((size_t)BL * DI * 4);
    unsigned short* xc_bf   = (unsigned short*)alloc((size_t)BL * DI * 2);
    float* xdbl             = (float*)alloc((size_t)BL * 48 * 4);
    float* delta            = (float*)alloc((size_t)BL * DI * 4);
    float* ybuf             = (float*)alloc((size_t)BL * DI * 4);
    unsigned short* yg_bf   = (unsigned short*)alloc((size_t)BL * DI * 2);
    float* pooled           = (float*)alloc((size_t)Bsz * DM * 4);
    (void)ws_size; (void)in_sizes; (void)n_in; (void)out_size;

    // -------- weight precision conversion (bf16 for WMMA) --------
    {
        int n0 = NLAY * 2 * DI * DM;
        cvt_f32_bf16_kernel<<<(n0 + 255) / 256, 256, 0, stream>>>(in_proj_w, w_in_bf, n0);
        int n1 = NLAY * 48 * DI;
        cvt_f32_bf16_kernel<<<(n1 + 255) / 256, 256, 0, stream>>>(x_proj_w, w_xp_bf, n1);
        int n2 = NLAY * DM * DI;
        cvt_f32_bf16_kernel<<<(n2 + 255) / 256, 256, 0, stream>>>(out_proj_w, w_op_bf, n2);
    }

    // -------- embedding --------
    embed_pos_kernel<<<(BL * DM + 255) / 256, 256, 0, stream>>>(tok, emb, pos_emb, hA);

    float* hcur  = hA;
    float* hnext = hB;
    for (int i = 0; i < NLAY; ++i) {
        // LayerNorm -> bf16 (one wave per token row)
        ln_rows_bf16_kernel<<<(BL * 32 + 255) / 256, 256, 0, stream>>>(
            hcur, ln_w + i * DM, ln_b + i * DM, hn_bf, BL);

        // in_proj: (BL,256) x (1024,256)^T -> xz (BL,1024), 16x64 strips
        {
            int waves = (BL / 16) * ((2 * DI) / 16 / 4);
            wmma_gemm_bf16_kernel<4><<<(waves + 3) / 4, 128, 0, stream>>>(
                hn_bf, w_in_bf + (size_t)i * 2 * DI * DM, nullptr, xz, BL, 2 * DI, DM);
        }

        // depthwise conv + SiLU
        conv_silu_kernel<<<(BL * DI + 255) / 256, 256, 0, stream>>>(
            xz, conv_w + (size_t)i * DI * DC, conv_b + (size_t)i * DI, xc, xc_bf);

        // x_proj: (BL,512) x (48,512)^T -> xdbl (BL,48), 16x48 strips
        {
            int waves = (BL / 16) * (48 / 16 / 3);
            wmma_gemm_bf16_kernel<3><<<(waves + 3) / 4, 128, 0, stream>>>(
                xc_bf, w_xp_bf + (size_t)i * 48 * DI, nullptr, xdbl, BL, 48, DI);
        }

        // delta
        dt_softplus_kernel<<<(BL * DI + 255) / 256, 256, 0, stream>>>(
            xdbl, dt_proj_w + (size_t)i * DI * DTR, dt_proj_b + (size_t)i * DI, delta);

        // selective scan: TDM-staged B/C strip in LDS (128 KB dynamic)
        scan_kernel<<<Bsz * (DI / 256), 256, (size_t)Lseq * 32 * 4, stream>>>(
            delta, xdbl, xc, A_log + (size_t)i * DI * DS, ybuf);

        // gate
        gate_kernel<<<(BL * DI + 255) / 256, 256, 0, stream>>>(
            ybuf, xc, Dp + (size_t)i * DI, xz, yg_bf);

        // out_proj + residual: (BL,512) x (256,512)^T + hcur -> hnext
        {
            int waves = (BL / 16) * (DM / 16 / 4);
            wmma_gemm_bf16_kernel<4><<<(waves + 3) / 4, 128, 0, stream>>>(
                yg_bf, w_op_bf + (size_t)i * DM * DI, hcur, hnext, BL, DM, DI);
        }
        float* tmp = hcur; hcur = hnext; hnext = tmp;
    }

    // pool + final LN + classifier
    pool_kernel<<<(Bsz * DM + 255) / 256, 256, 0, stream>>>(hcur, mask, pooled);
    final_head_kernel<<<Bsz, DM, 0, stream>>>(pooled, norm_w, norm_b, cls_w, cls_b, out);
}
